// EGeoGNNBlock_28080496181846
// MI455X (gfx1250) — matrix operand order repaired and checked by reading, not compiled, not run on hardware
//
#include <hip/hip_runtime.h>
#include <hip/hip_bf16.h>
#include <stdint.h>
#include <stddef.h>

#define G_    768
#define NA_   (G_*32)
#define NB_   (G_*64)
#define NANG_ (G_*128)
#define EADG_ (G_*192)
#define H_    128
#define HID_  512

typedef __attribute__((ext_vector_type(16))) __bf16 v16bf;
typedef __attribute__((ext_vector_type(8)))  float  v8f;

// ---------------- bf16 helpers (round-to-nearest-even) ----------------
static __device__ __forceinline__ unsigned short f2bfbits(float f) {
  unsigned u = __float_as_uint(f);
  u += 0x7FFFu + ((u >> 16) & 1u);
  return (unsigned short)(u >> 16);
}
static __device__ __forceinline__ __bf16 bits2bf(unsigned short h) {
  __bf16 r;
  __builtin_memcpy(&r, &h, 2);
  return r;
}
static __device__ __forceinline__ __bf16 f2bf(float f) { return bits2bf(f2bfbits(f)); }

// ---------------- utility kernels ----------------
__global__ void k_fill_u32(unsigned* p, unsigned v, long n) {
  long i = (long)blockIdx.x * blockDim.x + threadIdx.x;
  if (i < n) p[i] = v;
}

// transpose-convert weights: W[k*N+n] (f32) -> Wt[n*K+k] (bf16 bits), contiguous writes
__global__ void k_conv_w_t(const float* __restrict__ W, unsigned short* __restrict__ Wt,
                           int K, int N) {
  long i = (long)blockIdx.x * blockDim.x + threadIdx.x;
  if (i >= (long)K * N) return;
  int n = (int)(i / K), k = (int)(i % K);
  Wt[i] = f2bfbits(W[(size_t)k * N + n]);
}

// w1s[k*2+h] = (1/8) * sum_{d<64} w1[k*128 + h*64 + d];  w1s[768+h] = (1/8)*sum b1
__global__ void k_w1s(const float* __restrict__ w1, const float* __restrict__ b1,
                      float* __restrict__ w1s) {
  int t = blockIdx.x * blockDim.x + threadIdx.x;
  if (t < 768) {
    int k = t >> 1, h = t & 1;
    float s = 0.f;
    for (int d = 0; d < 64; ++d) s += w1[k * 128 + h * 64 + d];
    w1s[t] = s * 0.125f;
  } else if (t < 770) {
    int h = t - 768;
    float s = 0.f;
    for (int d = 0; d < 64; ++d) s += b1[h * 64 + d];
    w1s[t] = s * 0.125f;
  }
}

// ---------------- attention: logits + segment softmax ----------------
static __device__ __forceinline__ unsigned fkey(float f) {
  unsigned u = __float_as_uint(f);
  return (u & 0x80000000u) ? ~u : (u | 0x80000000u);
}
static __device__ __forceinline__ float funkey(unsigned k) {
  unsigned u = (k & 0x80000000u) ? (k & 0x7FFFFFFFu) : ~k;
  return __uint_as_float(u);
}

__global__ void k_logits(const float* __restrict__ nf, const float* __restrict__ ea,
                         const int* __restrict__ idxQ, const int* __restrict__ idxK,
                         const int* __restrict__ idxS, const float* __restrict__ w1s,
                         float* __restrict__ logits, unsigned* __restrict__ segmax, int E) {
  int t = blockIdx.x * blockDim.x + threadIdx.x;
  if (t >= E * 2) return;
  int e = t >> 1, h = t & 1;
  const float* q  = nf + (size_t)idxQ[e] * H_;
  const float* kk = nf + (size_t)idxK[e] * H_;
  const float* ee = ea + (size_t)e * H_;
  float s = w1s[768 + h];
#pragma unroll 4
  for (int d = 0; d < H_; ++d) {
    s += q[d]  * w1s[d * 2 + h]
       + kk[d] * w1s[(128 + d) * 2 + h]
       + ee[d] * w1s[(256 + d) * 2 + h];
  }
  logits[t] = s;
  atomicMax(&segmax[idxS[e] * 2 + h], fkey(s));
}

__global__ void k_expsum(const int* __restrict__ idxS, const float* __restrict__ logits,
                         const unsigned* __restrict__ segmax, float* __restrict__ ex,
                         float* __restrict__ segsum, int E) {
  int t = blockIdx.x * blockDim.x + threadIdx.x;
  if (t >= E * 2) return;
  int e = t >> 1, h = t & 1;
  float m = funkey(segmax[idxS[e] * 2 + h]);
  float v = __expf(logits[t] - m);
  ex[t] = v;
  atomicAdd(&segsum[idxS[e] * 2 + h], v);
}

__global__ void k_scatter(const int* __restrict__ idxS, const float* __restrict__ ex,
                          const float* __restrict__ segsum, const float* __restrict__ v,
                          float* __restrict__ outb, int E) {
  long t = (long)blockIdx.x * blockDim.x + threadIdx.x;
  if (t >= (long)E * H_) return;
  int e = (int)(t >> 7), f = (int)(t & 127), h = f >> 6;
  int s = idxS[e];
  float a = ex[e * 2 + h] / (segsum[s * 2 + h] + 1e-16f);
  atomicAdd(&outb[(size_t)s * H_ + f], a * v[t]);
}

// ---------------- WMMA GEMM over concatenated (optionally gathered) A ----------------
// Pieces are either: NP==1 single piece of width K, or NP pieces of width 128 each.
struct AArgs {
  const float* p[4];
  const int*   idx[4];
  int          np;
};

// C[M,N] = A[M,K] @ Wt^T + bias.  Wt bf16 [N,K] pre-transposed.
// grid = (M/128, N/128), block = 256 (8 waves); wave w owns rows m0..m0+15,
// 8 accumulators covering a 16x128 strip. B panel double-buffered in LDS.
template <int K, int NP>
__global__ void __launch_bounds__(256)
k_gemm_bf16(AArgs a, const unsigned short* __restrict__ Wt,
            const float* __restrict__ bias, float* __restrict__ C, int N) {
  __shared__ __align__(32) unsigned short Bsh[2][128 * 32]; // 2 x 8 KB
  int tid  = threadIdx.x;
  int wave = tid >> 5, lane = tid & 31;
  int half = lane >> 4, lm = lane & 15;
  int m0 = blockIdx.x * 128 + wave * 16;
  int n0 = blockIdx.y * 128;
  int rowA = m0 + lm;

  // per-lane row base pointers, gather resolved once
  const float* base[NP];
#pragma unroll
  for (int j = 0; j < NP; ++j) {
    const int w = (NP == 1) ? K : 128;
    int r = a.idx[j] ? a.idx[j][rowA] : rowA;
    base[j] = a.p[j] + (size_t)r * w;
  }

  v8f zero = {};
  v8f acc[8];
#pragma unroll
  for (int nt = 0; nt < 8; ++nt) acc[nt] = zero;

  // B staging: thread t owns 16 contiguous bf16 (col = t/2, khalf = t&1)
  int scol = tid >> 1, shalf = tid & 1;
  const unsigned short* wsrc = Wt + (size_t)(n0 + scol) * K + shalf * 16;
  const int sidx = scol * 32 + shalf * 16;

  constexpr int T = K / 32;

  // prologue: stage tile 0
  *(v16bf*)&Bsh[0][sidx] = *(const v16bf*)wsrc;
  __syncthreads();

#pragma unroll
  for (int kt = 0; kt < T; ++kt) {
    const int cur = kt & 1;

    // prefetch next B tile from global into registers (overlaps WMMA below)
    v16bf nxt;
    if (kt + 1 < T) nxt = *(const v16bf*)(wsrc + (kt + 1) * 32);

    // select A piece (branchless; widths are 128 when NP > 1)
    const float* bj;
    if (NP == 1) bj = base[0];
    else {
      int pc = kt >> 2;
      bj = base[0];
#pragma unroll
      for (int j = 1; j < NP; ++j) if (pc == j) bj = base[j];
    }
    const int kin = ((NP == 1) ? kt * 32 : (kt & 3) * 32) + half * 16;

    // A fragment: two aligned 32B vector loads + in-register bf16 convert
    v8f lo = *(const v8f*)(bj + kin);
    v8f hi = *(const v8f*)(bj + kin + 8);
    v16bf A;
#pragma unroll
    for (int q = 0; q < 4; ++q) {
      A[2 * q]     = f2bf(lo[2 * q]);
      A[2 * q + 1] = f2bf(lo[2 * q + 1]);
      A[2 * (q + 4)]     = f2bf(hi[2 * q]);
      A[2 * (q + 4) + 1] = f2bf(hi[2 * q + 1]);
    }

#pragma unroll
    for (int nt = 0; nt < 8; ++nt) {
      v16bf B = *(const v16bf*)&Bsh[cur][(nt * 16 + lm) * 32 + half * 16];
      acc[nt] = __builtin_amdgcn_wmma_f32_16x16x32_bf16(false, A, false, B,
                                                        (short)0, acc[nt], false, false);
    }

    if (kt + 1 < T) *(v16bf*)&Bsh[cur ^ 1][sidx] = nxt;
    __syncthreads();
  }

  // C layout: col = lm, row = half*8 + r
  int mr = m0 + half * 8;
#pragma unroll
  for (int nt = 0; nt < 8; ++nt) {
    int gc = n0 + nt * 16 + lm;
    float bb = bias[gc];
#pragma unroll
    for (int r = 0; r < 8; ++r)
      C[(size_t)(mr + r) * N + gc] = acc[nt][r] + bb;
  }
}

// ---------------- LayerNorm + ReLU (in place), one wave per row ----------------
__global__ void __launch_bounds__(256)
k_ln_relu(float* __restrict__ X, const float* __restrict__ g,
          const float* __restrict__ beta, int M, int N) {
  int wave = threadIdx.x >> 5, lane = threadIdx.x & 31;
  int row = blockIdx.x * 8 + wave;
  if (row >= M) return;
  float* x = X + (size_t)row * N;
  float s = 0.f, s2 = 0.f;
  for (int c = lane; c < N; c += 32) { float v = x[c]; s += v; s2 += v * v; }
#pragma unroll
  for (int o = 16; o > 0; o >>= 1) {
    s  += __shfl_xor(s, o, 32);
    s2 += __shfl_xor(s2, o, 32);
  }
  float mu  = s / N;
  float var = s2 / N - mu * mu;
  float inv = rsqrtf(var + 1e-5f);
  for (int c = lane; c < N; c += 32) {
    float v = (x[c] - mu) * inv * g[c] + beta[c];
    x[c] = v > 0.f ? v : 0.f;
  }
}

// ---------------- segment mean ----------------
__global__ void k_segsum(const float* __restrict__ x, const int* __restrict__ batch,
                         float* __restrict__ sum, float* __restrict__ cnt, int Nn) {
  long t = (long)blockIdx.x * blockDim.x + threadIdx.x;
  if (t >= (long)Nn * H_) return;
  int n = (int)(t >> 7), f = (int)(t & 127);
  int b = batch[n];
  atomicAdd(&sum[(size_t)b * H_ + f], x[t]);
  if (f == 0) atomicAdd(&cnt[b], 1.0f);
}

__global__ void k_divcnt(float* __restrict__ sum, const float* __restrict__ cnt, int Gn) {
  int t = blockIdx.x * blockDim.x + threadIdx.x;
  if (t >= Gn * H_) return;
  float c = cnt[t >> 7];
  sum[t] = sum[t] / (c < 1.f ? 1.f : c);
}

// ---------------- host orchestration ----------------
extern "C" void kernel_launch(void* const* d_in, const int* in_sizes, int n_in,
                              void* d_out, int out_size, void* d_ws, size_t ws_size,
                              hipStream_t stream) {
  (void)in_sizes; (void)n_in; (void)out_size; (void)ws_size;

  // params pytree (dict keys sorted): angle(16), atom(16), bond(16), glob(12)
  const float* P[60];
  for (int i = 0; i < 60; ++i) P[i] = (const float*)d_in[i];
  const float* atom_attr  = (const float*)d_in[60];
  const float* bond_attr  = (const float*)d_in[61];
  const float* angle_attr = (const float*)d_in[62];
  const float* dih_attr   = (const float*)d_in[63];
  const float* u_in       = (const float*)d_in[64];
  const int* ab_edges     = (const int*)d_in[65];
  const int* ba_edges     = (const int*)d_in[66];
  const int* ad_edges     = (const int*)d_in[67];
  const int* atom_batch   = (const int*)d_in[71];
  const int* bond_batch   = (const int*)d_in[72];
  const int* angle_batch  = (const int*)d_in[73];

  float* out_atom  = (float*)d_out;
  float* out_bond  = out_atom  + (size_t)NA_  * H_;
  float* out_angle = out_bond  + (size_t)NB_  * H_;
  float* out_u     = out_angle + (size_t)NANG_ * H_;

  // ---- workspace bump allocator ----
  size_t off = 0;
  auto alloc = [&](size_t bytes) -> void* {
    off = (off + 255) & ~(size_t)255;
    void* p = (char*)d_ws + off;
    off += bytes;
    return p;
  };

  unsigned short *wt_w2[3], *wt_m0[3], *wt_m1[3], *wt_m2[3];
  float* w1s[3];
  for (int i = 0; i < 3; ++i) {
    wt_w2[i] = (unsigned short*)alloc((size_t)256 * 128 * 2);
    wt_m0[i] = (unsigned short*)alloc((size_t)512 * 512 * 2);
    wt_m1[i] = (unsigned short*)alloc((size_t)512 * 512 * 2);
    wt_m2[i] = (unsigned short*)alloc((size_t)512 * 128 * 2);
    w1s[i]   = (float*)alloc(770 * 4);
  }
  unsigned short* wt_g0 = (unsigned short*)alloc((size_t)512 * 512 * 2);
  unsigned short* wt_g1 = (unsigned short*)alloc((size_t)512 * 512 * 2);
  unsigned short* wt_g2 = (unsigned short*)alloc((size_t)512 * 128 * 2);
  float* means = (float*)alloc((size_t)3 * G_ * H_ * 4);
  float* cnts  = (float*)alloc((size_t)3 * G_ * 4);
  float* gh1   = (float*)alloc((size_t)G_ * HID_ * 4);
  float* gh2   = (float*)alloc((size_t)G_ * HID_ * 4);
  unsigned* segmax = (unsigned*)alloc((size_t)NANG_ * 2 * 4);
  float*    segsum = (float*)alloc((size_t)NANG_ * 2 * 4);
  float*    logits = (float*)alloc((size_t)EADG_ * 2 * 4);
  float*    exb    = (float*)alloc((size_t)EADG_ * 2 * 4);
  float* regionB = (float*)alloc((size_t)NANG_ * HID_ * 4); // sent|recv|v, later h2
  float* h1      = (float*)alloc((size_t)NANG_ * HID_ * 4);

  auto fill = [&](void* p, unsigned v, long n) {
    k_fill_u32<<<dim3((unsigned)((n + 255) / 256)), 256, 0, stream>>>((unsigned*)p, v, n);
  };
  auto conv = [&](const float* W, unsigned short* Wt, int K, int N) {
    long n = (long)K * N;
    k_conv_w_t<<<dim3((unsigned)((n + 255) / 256)), 256, 0, stream>>>(W, Wt, K, N);
  };
  auto gemm = [&](const AArgs& a, const unsigned short* Wt, const float* bias,
                  float* C, int M, int N, int K) {
    dim3 grid((unsigned)(M / 128), (unsigned)(N / 128));
    if (K == 256 && a.np == 2)
      k_gemm_bf16<256, 2><<<grid, 256, 0, stream>>>(a, Wt, bias, C, N);
    else if (K == 512 && a.np == 4)
      k_gemm_bf16<512, 4><<<grid, 256, 0, stream>>>(a, Wt, bias, C, N);
    else
      k_gemm_bf16<512, 1><<<grid, 256, 0, stream>>>(a, Wt, bias, C, N);
  };

  struct Enc {
    int pb;                 // param base index
    const float *nf, *ea;
    const int *edges, *batch;
    int Nn, E;
    float* outp;
  };
  Enc encs[3] = {
    {16, atom_attr,  bond_attr,  ab_edges, atom_batch,  NA_,   NB_,   out_atom},
    {32, bond_attr,  angle_attr, ba_edges, bond_batch,  NB_,   NANG_, out_bond},
    { 0, angle_attr, dih_attr,   ad_edges, angle_batch, NANG_, EADG_, out_angle},
  };

  // ---- weight prep ----
  for (int i = 0; i < 3; ++i) {
    int pb = encs[i].pb;
    conv(P[pb + 3],  wt_w2[i], 256, 128);   // attn w2
    conv(P[pb + 7],  wt_m0[i], 512, 512);   // mlp0 w
    conv(P[pb + 11], wt_m1[i], 512, 512);   // mlp1 w
    conv(P[pb + 15], wt_m2[i], 512, 128);   // mlp2 w
    k_w1s<<<4, 256, 0, stream>>>(P[pb + 2], P[pb + 0], w1s[i]);
  }
  conv(P[51], wt_g0, 512, 512);
  conv(P[55], wt_g1, 512, 512);
  conv(P[59], wt_g2, 512, 128);

  // ---- encoders ----
  for (int i = 0; i < 3; ++i) {
    const Enc& E_ = encs[i];
    const int* row = E_.edges;
    const int* col = E_.edges + E_.E;
    float* sent = regionB;
    float* recv = regionB + (size_t)E_.Nn * H_;
    float* vbuf = regionB + (size_t)2 * E_.Nn * H_;
    float* h2   = regionB;

    fill(sent, 0u, (long)E_.Nn * H_);
    fill(recv, 0u, (long)E_.Nn * H_);

    for (int dir = 0; dir < 2; ++dir) {
      const int* idxQ = dir == 0 ? row : col;
      const int* idxK = dir == 0 ? col : row;
      const int* idxS = idxQ;
      float* outb = dir == 0 ? sent : recv;

      fill(segmax, 0u, (long)E_.Nn * 2);
      fill(segsum, 0u, (long)E_.Nn * 2);

      long nt = (long)E_.E * 2;
      k_logits<<<dim3((unsigned)((nt + 255) / 256)), 256, 0, stream>>>(
          E_.nf, E_.ea, idxQ, idxK, idxS, w1s[i], logits, segmax, E_.E);
      k_expsum<<<dim3((unsigned)((nt + 255) / 256)), 256, 0, stream>>>(
          idxS, logits, segmax, exb, segsum, E_.E);

      AArgs av = {};
      av.p[0] = E_.nf; av.idx[0] = idxK;
      av.p[1] = E_.ea; av.idx[1] = nullptr;
      av.np = 2;
      gemm(av, wt_w2[i], P[E_.pb + 1], vbuf, E_.E, H_, 2 * H_);

      long ns = (long)E_.E * H_;
      k_scatter<<<dim3((unsigned)((ns + 255) / 256)), 256, 0, stream>>>(
          idxS, exb, segsum, vbuf, outb, E_.E);
    }

    // MLP: [nf | sent | recv | u[batch]] (K=512) -> h1 -> h2 -> out
    AArgs a1 = {};
    a1.p[0] = E_.nf; a1.idx[0] = nullptr;
    a1.p[1] = sent;  a1.idx[1] = nullptr;
    a1.p[2] = recv;  a1.idx[2] = nullptr;
    a1.p[3] = u_in;  a1.idx[3] = E_.batch;
    a1.np = 4;
    gemm(a1, wt_m0[i], P[E_.pb + 4], h1, E_.Nn, HID_, 4 * H_);
    k_ln_relu<<<dim3((unsigned)(E_.Nn / 8)), 256, 0, stream>>>(
        h1, P[E_.pb + 6], P[E_.pb + 5], E_.Nn, HID_);

    AArgs a2 = {};
    a2.p[0] = h1; a2.idx[0] = nullptr; a2.np = 1;
    gemm(a2, wt_m1[i], P[E_.pb + 8], h2, E_.Nn, HID_, HID_);
    k_ln_relu<<<dim3((unsigned)(E_.Nn / 8)), 256, 0, stream>>>(
        h2, P[E_.pb + 10], P[E_.pb + 9], E_.Nn, HID_);

    AArgs a3 = {};
    a3.p[0] = h2; a3.idx[0] = nullptr; a3.np = 1;
    gemm(a3, wt_m2[i], P[E_.pb + 12], E_.outp, E_.Nn, H_, HID_);
  }

  // ---- global: segment means over encoder outputs ----
  struct Seg { const float* x; const int* batch; int Nn; };
  Seg segs[3] = {
    {out_atom,  atom_batch,  NA_},
    {out_bond,  bond_batch,  NB_},
    {out_angle, angle_batch, NANG_},
  };
  fill(means, 0u, (long)3 * G_ * H_);
  fill(cnts,  0u, (long)3 * G_);
  for (int i = 0; i < 3; ++i) {
    float* mn = means + (size_t)i * G_ * H_;
    float* cn = cnts + (size_t)i * G_;
    long n = (long)segs[i].Nn * H_;
    k_segsum<<<dim3((unsigned)((n + 255) / 256)), 256, 0, stream>>>(
        segs[i].x, segs[i].batch, mn, cn, segs[i].Nn);
    k_divcnt<<<dim3((unsigned)((G_ * H_ + 255) / 256)), 256, 0, stream>>>(mn, cn, G_);
  }

  // ---- global MLP: [u | meanA | meanB | meanC] -> 512 -> 512 -> 128 ----
  AArgs ag = {};
  ag.p[0] = u_in;                        ag.idx[0] = nullptr;
  ag.p[1] = means;                       ag.idx[1] = nullptr;
  ag.p[2] = means + (size_t)G_ * H_;     ag.idx[2] = nullptr;
  ag.p[3] = means + (size_t)2 * G_ * H_; ag.idx[3] = nullptr;
  ag.np = 4;
  gemm(ag, wt_g0, P[48], gh1, G_, HID_, 4 * H_);
  k_ln_relu<<<dim3((unsigned)(G_ / 8)), 256, 0, stream>>>(gh1, P[50], P[49], G_, HID_);

  AArgs ag2 = {};
  ag2.p[0] = gh1; ag2.idx[0] = nullptr; ag2.np = 1;
  gemm(ag2, wt_g1, P[52], gh2, G_, HID_, HID_);
  k_ln_relu<<<dim3((unsigned)(G_ / 8)), 256, 0, stream>>>(gh2, P[54], P[53], G_, HID_);

  AArgs ag3 = {};
  ag3.p[0] = gh2; ag3.idx[0] = nullptr; ag3.np = 1;
  gemm(ag3, wt_g2, P[56], out_u, G_, H_, HID_);
}
